// AxialShift_85100482003616
// MI455X (gfx1250) — matrix operand order repaired and checked by reading, not compiled, not run on hardware
//
#include <hip/hip_runtime.h>
#include <math.h>

// ---------------------------------------------------------------------------
// AxialShift block on MI455X (gfx1250): 5 channel-GEMMs via v_wmma_f32_16x16x32_f16,
// shifts folded into LDS B-tile staging (channel chunks of 32 == K steps),
// GroupNorm via deterministic two-stage reductions.
// ---------------------------------------------------------------------------

typedef __attribute__((ext_vector_type(16))) _Float16 v16h;
typedef __attribute__((ext_vector_type(8)))  _Float16 v8h;
typedef __attribute__((ext_vector_type(8)))  float    v8f;

#define C_DIM   224
#define VOX     32768          // 32*32*32
#define NB      2
#define KCH     7              // 224 / 32
#define WN      50176          // 224*224
#define EPS_GN  1e-5f
#define LPITCH  40             // LDS B-tile row pitch in halves (32 + 8 pad)

__device__ __forceinline__ float gelu_exact(float x) {
    return 0.5f * x * (1.0f + erff(x * 0.70710678118654752f));
}

// A fragment (16x32 f16): lane holds row M = lane%16; halves K in
// [off,off+8) and [off+16,off+24), off = (lane<16)?0:8. Row is K-contiguous
// in the f16 weight matrix, so load straight from global (L2-resident).
__device__ __forceinline__ v16h load_a_frag(const _Float16* __restrict__ wrow, int lane) {
    const int off = (lane < 16) ? 0 : 8;
    const v8h lo = *(const v8h*)(wrow + off);
    const v8h hi = *(const v8h*)(wrow + off + 16);
    v16h a;
#pragma unroll
    for (int i = 0; i < 8; ++i) { a[i] = lo[i]; a[i + 8] = hi[i]; }
    return a;
}

// B fragment (32x16 f16): lane holds col N = lane%16; halves K in
// [koff,koff+8) and [koff+16,koff+24), koff = (lane<16)?0:8. LDS tile is [n][k] with pitch 40.
__device__ __forceinline__ v16h load_b_frag(const _Float16* lds, int lane, int nw) {
    const int n    = nw + (lane & 15);
    const int koff = (lane < 16) ? 0 : 8;
    const _Float16* p = lds + n * LPITCH + koff;
    const v8h lo = *(const v8h*)(p);
    const v8h hi = *(const v8h*)(p + 16);
    v16h b;
#pragma unroll
    for (int i = 0; i < 8; ++i) { b[i] = lo[i]; b[i + 8] = hi[i]; }
    return b;
}

// ------------------------- weight conversion -------------------------------
__global__ __launch_bounds__(256)
void k_cvt_w(const float* __restrict__ w1, const float* __restrict__ w21,
             const float* __restrict__ w22, const float* __restrict__ w23,
             const float* __restrict__ w3, _Float16* __restrict__ dst) {
    int i = blockIdx.x * 256 + threadIdx.x;          // 5*50176 = 980*256 exact
    int m = i / WN, r = i % WN;
    const float* src = (m == 0) ? w1 : (m == 1) ? w21 : (m == 2) ? w22 : (m == 3) ? w23 : w3;
    dst[i] = (_Float16)src[r];
}

// ------------------------- GEMM1: t = W1*x + b1 ----------------------------
__global__ __launch_bounds__(256)
void k_gemm1(const float* __restrict__ x, const _Float16* __restrict__ W1,
             const float* __restrict__ b1, float* __restrict__ t) {
    __shared__ _Float16 ldsB[64 * LPITCH];
    const int tid  = threadIdx.x, lane = tid & 31, wave = tid >> 5;
    const int b    = blockIdx.z;
    const int m0   = (blockIdx.y * 2 + (wave >> 2)) * 16;
    const int nblk = blockIdx.x * 64;
    const int nw   = (wave & 3) * 16;
    const int mrow = m0 + (lane & 15);
    const size_t baseIn = (size_t)b * C_DIM * VOX;

    const int sn = tid & 63;        // staging: voxel within tile
    const int kq = tid >> 6;        // staging: k-group 0..3

    v8f acc = {};
    for (int kb = 0; kb < KCH; ++kb) {
        __syncthreads();
#pragma unroll
        for (int i = 0; i < 8; ++i) {
            int k = kq * 8 + i;
            float v = x[baseIn + (size_t)(kb * 32 + k) * VOX + nblk + sn];
            ldsB[sn * LPITCH + k] = (_Float16)v;
        }
        __syncthreads();
        v16h a = load_a_frag(W1 + (size_t)mrow * C_DIM + kb * 32, lane);
        v16h bf = load_b_frag(ldsB, lane, nw);
        acc = __builtin_amdgcn_wmma_f32_16x16x32_f16(false, a, false, bf,
                                                     (short)0, acc, false, false);
    }
    const int nout = nblk + nw + (lane & 15);
    const int mhi  = (lane >> 4) * 8;
#pragma unroll
    for (int r = 0; r < 8; ++r) {
        int m = m0 + mhi + r;
        t[baseIn + (size_t)m * VOX + nout] = acc[r] + b1[m];
    }
}

// ---------------- deterministic two-stage GN statistics --------------------
__global__ __launch_bounds__(256)
void k_reduce(const float* __restrict__ src, float* __restrict__ partials) {
    __shared__ float s1[256], s2[256];
    const int b = blockIdx.y;
    const size_t n = (size_t)C_DIM * VOX;
    const float* p = src + (size_t)b * n;
    float sum = 0.f, sq = 0.f;
    for (size_t i = (size_t)blockIdx.x * 256 + threadIdx.x; i < n; i += 256 * 256) {
        float v = p[i]; sum += v; sq += v * v;
    }
    s1[threadIdx.x] = sum; s2[threadIdx.x] = sq;
    __syncthreads();
    for (int s = 128; s > 0; s >>= 1) {
        if (threadIdx.x < s) {
            s1[threadIdx.x] += s1[threadIdx.x + s];
            s2[threadIdx.x] += s2[threadIdx.x + s];
        }
        __syncthreads();
    }
    if (threadIdx.x == 0) {
        partials[(b * 256 + blockIdx.x) * 2 + 0] = s1[0];
        partials[(b * 256 + blockIdx.x) * 2 + 1] = s2[0];
    }
}

__global__ void k_finalize(const float* __restrict__ partials, float* __restrict__ muinv) {
    int b = threadIdx.x;
    if (b >= NB) return;
    float sum = 0.f, sq = 0.f;
    for (int i = 0; i < 256; ++i) {            // fixed order -> deterministic
        sum += partials[(b * 256 + i) * 2 + 0];
        sq  += partials[(b * 256 + i) * 2 + 1];
    }
    float n   = (float)((size_t)C_DIM * VOX);
    float mu  = sum / n;
    float var = sq / n - mu * mu;
    muinv[b * 2 + 0] = mu;
    muinv[b * 2 + 1] = rsqrtf(var + EPS_GN);
}

// ---------------- h = f16( gelu( gn1(t) ) ) --------------------------------
__global__ __launch_bounds__(256)
void k_gn_gelu(const float* __restrict__ t, const float* __restrict__ muinv,
               const float* __restrict__ g, const float* __restrict__ beta,
               _Float16* __restrict__ h16) {
    size_t i = (size_t)blockIdx.x * 256 + threadIdx.x;   // exact grid
    int b = (int)(i / ((size_t)C_DIM * VOX));
    int c = (int)((i / VOX) % C_DIM);
    float v = (t[i] - muinv[b * 2]) * muinv[b * 2 + 1] * g[c] + beta[c];
    h16[i] = (_Float16)gelu_exact(v);
}

// ------- fused axial GEMMs: S = gelu(W21*h_d+b21)+gelu(W22*h_h+b22)+gelu(W23*h_w+b23)
__global__ __launch_bounds__(256)
void k_gemm234(const _Float16* __restrict__ h16,
               const _Float16* __restrict__ W21, const _Float16* __restrict__ W22,
               const _Float16* __restrict__ W23,
               const float* __restrict__ b21, const float* __restrict__ b22,
               const float* __restrict__ b23, float* __restrict__ S) {
    __shared__ _Float16 ldsB[64 * LPITCH];
    const int tid  = threadIdx.x, lane = tid & 31, wave = tid >> 5;
    const int b    = blockIdx.z;
    const int m0   = (blockIdx.y * 2 + (wave >> 2)) * 16;
    const int nblk = blockIdx.x * 64;
    const int nw   = (wave & 3) * 16;
    const int mrow = m0 + (lane & 15);
    const size_t baseIn = (size_t)b * C_DIM * VOX;

    const int sn = tid & 63;
    const int kq = tid >> 6;
    const int vi = nblk + sn;
    const int vd = vi >> 10, vh = (vi >> 5) & 31, vw = vi & 31;

    const _Float16* Ws[3] = {W21, W22, W23};
    v8f acc[3] = {v8f{}, v8f{}, v8f{}};

    for (int kb = 0; kb < KCH; ++kb) {
        const int s = kb - 3;   // channel-chunk shift (chunk == K block of 32)
#pragma unroll
        for (int j = 0; j < 3; ++j) {
            int sd, sh, sw;
            if (j == 0)      { sd = vd - s; sh = vh;         sw = vw;         }
            else if (j == 1) { sd = vd - 3; sh = vh - s - 3; sw = vw - 3;     }
            else             { sd = vd - 6; sh = vh - 6;     sw = vw - s - 6; }
            const bool ok = ((unsigned)sd < 32u) & ((unsigned)sh < 32u) & ((unsigned)sw < 32u);
            const size_t svox = (size_t)(sd * 1024 + sh * 32 + sw);
            __syncthreads();
#pragma unroll
            for (int i = 0; i < 8; ++i) {
                int k = kq * 8 + i;
                _Float16 v = ok ? h16[baseIn + (size_t)(kb * 32 + k) * VOX + svox]
                                : (_Float16)0.0f;
                ldsB[sn * LPITCH + k] = v;
            }
            __syncthreads();
            v16h a  = load_a_frag(Ws[j] + (size_t)mrow * C_DIM + kb * 32, lane);
            v16h bf = load_b_frag(ldsB, lane, nw);
            acc[j] = __builtin_amdgcn_wmma_f32_16x16x32_f16(false, a, false, bf,
                                                            (short)0, acc[j], false, false);
        }
    }
    const int nout = nblk + nw + (lane & 15);
    const int mhi  = (lane >> 4) * 8;
#pragma unroll
    for (int r = 0; r < 8; ++r) {
        int m = m0 + mhi + r;
        float v = gelu_exact(acc[0][r] + b21[m])
                + gelu_exact(acc[1][r] + b22[m])
                + gelu_exact(acc[2][r] + b23[m]);
        S[baseIn + (size_t)m * VOX + nout] = v;
    }
}

// -------- out = W3 * gn2(S) + b3 + x  (GN2 affine folded into staging) -----
__global__ __launch_bounds__(256)
void k_gemm3(const float* __restrict__ S, const _Float16* __restrict__ W3,
             const float* __restrict__ b3, const float* __restrict__ x,
             const float* __restrict__ muinv2, const float* __restrict__ g2,
             const float* __restrict__ beta2, float* __restrict__ out) {
    __shared__ _Float16 ldsB[64 * LPITCH];
    const int tid  = threadIdx.x, lane = tid & 31, wave = tid >> 5;
    const int b    = blockIdx.z;
    const int m0   = (blockIdx.y * 2 + (wave >> 2)) * 16;
    const int nblk = blockIdx.x * 64;
    const int nw   = (wave & 3) * 16;
    const int mrow = m0 + (lane & 15);
    const size_t baseIn = (size_t)b * C_DIM * VOX;

    const float mu  = muinv2[b * 2 + 0];
    const float inv = muinv2[b * 2 + 1];

    const int sn = tid & 63;
    const int kq = tid >> 6;

    v8f acc = {};
    for (int kb = 0; kb < KCH; ++kb) {
        __syncthreads();
#pragma unroll
        for (int i = 0; i < 8; ++i) {
            int k = kq * 8 + i;
            int c = kb * 32 + k;
            float scale = inv * g2[c];
            float off   = beta2[c] - mu * scale;
            float v = S[baseIn + (size_t)c * VOX + nblk + sn] * scale + off;
            ldsB[sn * LPITCH + k] = (_Float16)v;
        }
        __syncthreads();
        v16h a  = load_a_frag(W3 + (size_t)mrow * C_DIM + kb * 32, lane);
        v16h bf = load_b_frag(ldsB, lane, nw);
        acc = __builtin_amdgcn_wmma_f32_16x16x32_f16(false, a, false, bf,
                                                     (short)0, acc, false, false);
    }
    const int nout = nblk + nw + (lane & 15);
    const int mhi  = (lane >> 4) * 8;
#pragma unroll
    for (int r = 0; r < 8; ++r) {
        int m = m0 + mhi + r;
        size_t idx = baseIn + (size_t)m * VOX + nout;
        out[idx] = acc[r] + b3[m] + x[idx];
    }
}

// ---------------------------------------------------------------------------
extern "C" void kernel_launch(void* const* d_in, const int* in_sizes, int n_in,
                              void* d_out, int out_size, void* d_ws, size_t ws_size,
                              hipStream_t stream) {
    const float* x   = (const float*)d_in[0];
    const float* w1  = (const float*)d_in[1];
    const float* b1  = (const float*)d_in[2];
    const float* w21 = (const float*)d_in[3];
    const float* b21 = (const float*)d_in[4];
    const float* w22 = (const float*)d_in[5];
    const float* b22 = (const float*)d_in[6];
    const float* w23 = (const float*)d_in[7];
    const float* b23 = (const float*)d_in[8];
    const float* w3  = (const float*)d_in[9];
    const float* b3  = (const float*)d_in[10];
    const float* g1  = (const float*)d_in[11];
    const float* bn1 = (const float*)d_in[12];
    const float* g2  = (const float*)d_in[13];
    const float* bn2 = (const float*)d_in[14];

    char* ws = (char*)d_ws;
    const size_t nBCV = (size_t)NB * C_DIM * VOX;           // 14,680,064
    _Float16* wf16     = (_Float16*)ws;                     // 5*WN halves  (~0.5 MB)
    float*    tS       = (float*)(ws + 512 * 1024);         // t, later S   (58.7 MB)
    _Float16* h16      = (_Float16*)(ws + 512 * 1024 + nBCV * 4);        // 29.4 MB
    float*    partials = (float*)(ws + 512 * 1024 + nBCV * 4 + nBCV * 2);
    float*    muinv1   = partials + NB * 256 * 2;
    float*    muinv2   = muinv1 + NB * 2;

    const dim3 gGemm(VOX / 64, C_DIM / 32, NB);             // (512, 7, 2)

    k_cvt_w<<<5 * WN / 256, 256, 0, stream>>>(w1, w21, w22, w23, w3, wf16);
    k_gemm1<<<gGemm, 256, 0, stream>>>(x, wf16, b1, tS);
    k_reduce<<<dim3(256, NB), 256, 0, stream>>>(tS, partials);
    k_finalize<<<1, NB, 0, stream>>>(partials, muinv1);
    k_gn_gelu<<<(unsigned)(nBCV / 256), 256, 0, stream>>>(tS, muinv1, g1, bn1, h16);
    k_gemm234<<<gGemm, 256, 0, stream>>>(h16, wf16 + WN, wf16 + 2 * WN, wf16 + 3 * WN,
                                         b21, b22, b23, tS);
    k_reduce<<<dim3(256, NB), 256, 0, stream>>>(tS, partials);
    k_finalize<<<1, NB, 0, stream>>>(partials, muinv2);
    k_gemm3<<<gGemm, 256, 0, stream>>>(tS, wf16 + 4 * WN, b3, x, muinv2, g2, bn2,
                                       (float*)d_out);
}
